// S4DLayer_43963285242185
// MI455X (gfx1250) — matrix operand order repaired and checked by reading, not compiled, not run on hardware
//
#include <hip/hip_runtime.h>
#include <hip/hip_bf16.h>

// Problem constants (from the reference)
#define B_  16
#define C_  256
#define S_  64
#define L_  4096
#define T_  64               // chunk length (scan block)
#define NCH (L_ / T_)        // 64 chunks

typedef __attribute__((ext_vector_type(16))) __bf16 v16bf;
typedef __attribute__((ext_vector_type(2)))  __bf16 v2bf;
typedef __attribute__((ext_vector_type(8)))  float  v8f;

union FragBF {
  v16bf v;
  unsigned int u[8];
};

__device__ __forceinline__ unsigned short f2bf(float f) {
  union { float f; unsigned int u; } x;
  x.f = f;
  unsigned int r = x.u + 0x7FFFu + ((x.u >> 16) & 1u);  // round-to-nearest-even
  return (unsigned short)(r >> 16);
}

// Pack two f32 -> two bf16 in one dword; prefer the native packed convert.
__device__ __forceinline__ unsigned int pack2bf(float lo, float hi) {
#if __has_builtin(__builtin_amdgcn_cvt_pk_bf16_f32)
  union { v2bf v; unsigned int u; } x;
  x.v = __builtin_amdgcn_cvt_pk_bf16_f32(lo, hi);
  return x.u;
#elif defined(__AMDGCN__)
  union { v2bf v; unsigned int u; } x;
  x.v.x = (__bf16)lo;
  x.v.y = (__bf16)hi;
  return x.u;
#else
  return (unsigned int)f2bf(lo) | ((unsigned int)f2bf(hi) << 16);
#endif
}

// A-matrix fragment (16x32 bf16), rows [rowBase..rowBase+15], row stride rs halves,
// K columns [kb..kb+31]. 16-bit A layout: lane<16 -> M=lane, K in {0..7,16..23};
// lanes 16..31 add +8 to K. Consecutive halves = contiguous K pairs -> dword loads.
__device__ __forceinline__ v16bf loadA(const unsigned short* base, int rowBase,
                                       int rs, int kb, int lane) {
  FragBF f;
  int g = (lane >> 4) & 1;
  int m = lane & 15;
  const unsigned short* row = base + (rowBase + m) * rs;
#pragma unroll
  for (int j = 0; j < 8; ++j) {
    int k = kb + ((j & 4) ? 16 : 0) + (j & 3) * 2 + g * 8;
    f.u[j] = *(const unsigned int*)(row + k);
  }
  return f.v;
}

// B-matrix fragment (32x16 bf16) from an n-major LDS matrix [16][rs]:
// lane<16 -> N=lane, K in {0..7,16..23}; lanes 16..31 -> N=lane-16, add +8 to K.
__device__ __forceinline__ v16bf loadB(const unsigned short* base, int rs,
                                       int kb, int lane) {
  FragBF f;
  int g = (lane >> 4) & 1;
  int n = lane & 15;
  const unsigned short* row = base + n * rs;
#pragma unroll
  for (int j = 0; j < 8; ++j) {
    int k = kb + ((j & 4) ? 16 : 0) + (j & 3) * 2 + g * 8;
    f.u[j] = *(const unsigned int*)(row + k);
  }
  return f.v;
}

__global__ __launch_bounds__(128)
void s4d_chunkscan_wmma(const float* __restrict__ xg,
                        const float* __restrict__ ldt,
                        const float* __restrict__ arl,
                        const float* __restrict__ aim,
                        const float* __restrict__ cin,
                        float* __restrict__ out) {
  // --- LDS (~48 KB; one channel per workgroup) ---
  // State rows are INTERLEAVED: row 2s = re, row 2s+1 = im  (s = 0..63)
  __shared__ unsigned short Ktoe[T_ * T_];        // [t][u] lower-tri Toeplitz of k
  __shared__ unsigned short Pm[T_ * 2 * S_];      // [t][2S] carry projection
  __shared__ unsigned short Qm[2 * S_ * T_];      // [2S][u] state gather
  __shared__ unsigned short Xb[B_ * T_];          // n-major x chunk (bf16)
  __shared__ unsigned short Hb[B_ * 2 * S_];      // n-major bf16 state (carry B input)
  __shared__ float sh_adr[S_], sh_adi[S_];
  __shared__ float sh_bcr[S_], sh_bci[S_];
  __shared__ float sh_wTr[S_], sh_wTi[S_];
  __shared__ float sh_k[T_];

  const int tid  = threadIdx.x;
  const int lane = tid & 31;
  const int w    = tid >> 5;       // wave id 0..3 (uniform per wave)
  const int ch   = blockIdx.x;     // channel
  const int g    = (lane >> 4) & 1;
  const int n    = lane & 15;

  // ---------- per-channel parameter setup ----------
  if (tid < S_) {
    int s = tid;
    float dt  = __expf(ldt[ch]);
    float ar  = -__expf(arl[ch * S_ + s]);
    float ai  = aim[ch * S_ + s];
    float adr = ar * dt, adi = ai * dt;
    sh_adr[s] = adr; sh_adi[s] = adi;
    // b_c = cc * (exp(a*dt) - 1) / a
    float e  = __expf(adr);
    float wr = e * __cosf(adi), wi = e * __sinf(adi);
    float nr = wr - 1.0f, ni = wi;
    float den = ar * ar + ai * ai;
    float fr = (nr * ar + ni * ai) / den;
    float fi = (ni * ar - nr * ai) / den;
    float ccr = cin[(ch * S_ + s) * 2 + 0];
    float cci = cin[(ch * S_ + s) * 2 + 1];
    sh_bcr[s] = ccr * fr - cci * fi;
    sh_bci[s] = ccr * fi + cci * fr;
    // w^T (chunk-to-chunk state decay)
    float eT = __expf((float)T_ * adr);
    float pT = (float)T_ * adi;
    sh_wTr[s] = eT * __cosf(pT);
    sh_wTi[s] = eT * __sinf(pT);
  }
  __syncthreads();

  // k[j] = 2 * sum_s Re(b_c[s] * w[s]^j)
  if (tid < T_) {
    float j = (float)tid;
    float acc = 0.0f;
    for (int s = 0; s < S_; ++s) {
      float e = __expf(j * sh_adr[s]);
      float p = j * sh_adi[s];
      float wr = e * __cosf(p), wi = e * __sinf(p);
      acc += 2.0f * (sh_bcr[s] * wr - sh_bci[s] * wi);
    }
    sh_k[tid] = acc;
  }
  __syncthreads();

  // Toeplitz K[t][u] = (u<=t) ? k[t-u] : 0
  for (int i = tid; i < T_ * T_; i += 128) {
    int r = i >> 6, cl = i & 63;
    Ktoe[i] = (cl <= r) ? f2bf(sh_k[r - cl]) : (unsigned short)0;
  }
  // P[t][2s] = 2Re(b w^{t+1}),  P[t][2s+1] = -2Im(b w^{t+1})
  for (int i = tid; i < T_ * 2 * S_; i += 128) {
    int t = i >> 7, col = i & 127, s = col >> 1;
    float j = (float)(t + 1);
    float e = __expf(j * sh_adr[s]);
    float p = j * sh_adi[s];
    float wr = e * __cosf(p), wi = e * __sinf(p);
    float rb = sh_bcr[s] * wr - sh_bci[s] * wi;
    float ib = sh_bcr[s] * wi + sh_bci[s] * wr;
    Pm[i] = f2bf((col & 1) ? -2.0f * ib : 2.0f * rb);
  }
  // Q[2s][u] = Re(w^{63-u}),  Q[2s+1][u] = Im(w^{63-u})
  for (int i = tid; i < 2 * S_ * T_; i += 128) {
    int r = i >> 6, u = i & 63, s = r >> 1;
    float j = (float)(T_ - 1 - u);
    float e = __expf(j * sh_adr[s]);
    float p = j * sh_adi[s];
    Qm[i] = f2bf((r & 1) ? e * __sinf(p) : e * __cosf(p));
  }
  // zero bf16 state
  for (int i = tid; i < B_ * 2 * S_; i += 128) Hb[i] = (unsigned short)0;

  // stage chunk 0 of x into Xb (n-major bf16); thread -> (b, 8 contiguous u)
  const int xb_b  = tid >> 3;
  const int xb_u0 = (tid & 7) * 8;
  const float* xsrc = xg + ((size_t)xb_b * C_ + ch) * L_ + xb_u0;
  {
    float4 v0 = ((const float4*)xsrc)[0];
    float4 v1 = ((const float4*)xsrc)[1];
    uint4 pk;
    pk.x = pack2bf(v0.x, v0.y); pk.y = pack2bf(v0.z, v0.w);
    pk.z = pack2bf(v1.x, v1.y); pk.w = pack2bf(v1.z, v1.w);
    *(uint4*)(Xb + xb_b * T_ + xb_u0) = pk;
  }
  __syncthreads();

  // ---------- hoist all loop-invariant A fragments into registers ----------
  v16bf ka0 = loadA(Ktoe, 16 * w, T_, 0, lane);
  v16bf ka1 = loadA(Ktoe, 16 * w, T_, 32, lane);
  v16bf pa0 = loadA(Pm, 16 * w, 2 * S_, 0, lane);
  v16bf pa1 = loadA(Pm, 16 * w, 2 * S_, 32, lane);
  v16bf pa2 = loadA(Pm, 16 * w, 2 * S_, 64, lane);
  v16bf pa3 = loadA(Pm, 16 * w, 2 * S_, 96, lane);
  v16bf qa00 = loadA(Qm, 16 * (2 * w + 0), T_, 0, lane);
  v16bf qa01 = loadA(Qm, 16 * (2 * w + 0), T_, 32, lane);
  v16bf qa10 = loadA(Qm, 16 * (2 * w + 1), T_, 0, lane);
  v16bf qa11 = loadA(Qm, 16 * (2 * w + 1), T_, 32, lane);

  // Per-lane w^T constants for the register-resident state update.
  // Tile ti covers state rows 16*(2w+ti)+r+8g; even/odd row pair -> s = 8*(2w+ti)+j+4g
  float wtr[2][4], wti[2][4];
#pragma unroll
  for (int ti = 0; ti < 2; ++ti)
#pragma unroll
    for (int j = 0; j < 4; ++j) {
      int s = 8 * (2 * w + ti) + j + 4 * g;
      wtr[ti][j] = sh_wTr[s];
      wti[ti][j] = sh_wTi[s];
    }

  // f32 master state, register resident: Hst[ti][r] -> row 16*(2w+ti)+r+8g, col n
  float Hst[2][8];
#pragma unroll
  for (int ti = 0; ti < 2; ++ti)
#pragma unroll
    for (int r = 0; r < 8; ++r) Hst[ti][r] = 0.0f;

  // ---------- chunked scan over L ----------
  for (int chunk = 0; chunk < NCH; ++chunk) {
    const int l0 = chunk * T_;

    // Early-issue next chunk's global x load (lands during the WMMA work)
    float4 nx0, nx1;
    const bool have_next = (chunk + 1 < NCH);
    if (have_next) {
      const float* src = xsrc + (size_t)(l0 + T_);
      nx0 = ((const float4*)src)[0];
      nx1 = ((const float4*)src)[1];
      if (chunk + 2 < NCH) __builtin_prefetch(src + T_, 0, 1);
    }

    // ---- B fragments for this chunk (Xb and Hb are valid here) ----
    v16bf xb0 = loadB(Xb, T_, 0, lane);
    v16bf xb1 = loadB(Xb, T_, 32, lane);
    v16bf hb0 = loadB(Hb, 2 * S_, 0, lane);
    v16bf hb1 = loadB(Hb, 2 * S_, 32, lane);
    v16bf hb2 = loadB(Hb, 2 * S_, 64, lane);
    v16bf hb3 = loadB(Hb, 2 * S_, 96, lane);

    // ---- Y tile: intra-chunk Toeplitz conv + cross-chunk carry ----
    v8f accY = {};
    accY = __builtin_amdgcn_wmma_f32_16x16x32_bf16(false, ka0, false, xb0, (short)0, accY, false, false);
    accY = __builtin_amdgcn_wmma_f32_16x16x32_bf16(false, ka1, false, xb1, (short)0, accY, false, false);
    accY = __builtin_amdgcn_wmma_f32_16x16x32_bf16(false, pa0, false, hb0, (short)0, accY, false, false);
    accY = __builtin_amdgcn_wmma_f32_16x16x32_bf16(false, pa1, false, hb1, (short)0, accY, false, false);
    accY = __builtin_amdgcn_wmma_f32_16x16x32_bf16(false, pa2, false, hb2, (short)0, accY, false, false);
    accY = __builtin_amdgcn_wmma_f32_16x16x32_bf16(false, pa3, false, hb3, (short)0, accY, false, false);

    // C/D layout: VGPR r -> M = r + 8g, N = n (= batch)
    {
      float* dst = out + ((size_t)n * C_ + ch) * L_ + l0 + 16 * w + 8 * g;
#pragma unroll
      for (int r = 0; r < 8; ++r) dst[r] = accY[r];
    }

    // ---- state gather V = Q @ X, then register-resident complex update ----
    v8f accV0 = {}, accV1 = {};
    accV0 = __builtin_amdgcn_wmma_f32_16x16x32_bf16(false, qa00, false, xb0, (short)0, accV0, false, false);
    accV0 = __builtin_amdgcn_wmma_f32_16x16x32_bf16(false, qa01, false, xb1, (short)0, accV0, false, false);
    accV1 = __builtin_amdgcn_wmma_f32_16x16x32_bf16(false, qa10, false, xb0, (short)0, accV1, false, false);
    accV1 = __builtin_amdgcn_wmma_f32_16x16x32_bf16(false, qa11, false, xb1, (short)0, accV1, false, false);

    // H <- w^T * H + V  (re/im are adjacent VGPRs of the same lane)
    uint4 hbw[2];
#pragma unroll
    for (int ti = 0; ti < 2; ++ti) {
      unsigned int pk[4];
#pragma unroll
      for (int j = 0; j < 4; ++j) {
        int r = 2 * j;
        float vr = (ti == 0) ? accV0[r] : accV1[r];
        float vi = (ti == 0) ? accV0[r + 1] : accV1[r + 1];
        float hr = Hst[ti][r], hi = Hst[ti][r + 1];
        float nr = wtr[ti][j] * hr - wti[ti][j] * hi + vr;
        float ni = wti[ti][j] * hr + wtr[ti][j] * hi + vi;
        Hst[ti][r] = nr;
        Hst[ti][r + 1] = ni;
        pk[j] = pack2bf(nr, ni);
      }
      hbw[ti].x = pk[0]; hbw[ti].y = pk[1]; hbw[ti].z = pk[2]; hbw[ti].w = pk[3];
    }

    __syncthreads();  // all reads of Xb/Hb for this chunk are complete

    // publish new bf16 state and stage next x chunk
#pragma unroll
    for (int ti = 0; ti < 2; ++ti)
      *(uint4*)(Hb + n * (2 * S_) + 16 * (2 * w + ti) + 8 * g) = hbw[ti];

    if (have_next) {
      uint4 pk;
      pk.x = pack2bf(nx0.x, nx0.y); pk.y = pack2bf(nx0.z, nx0.w);
      pk.z = pack2bf(nx1.x, nx1.y); pk.w = pack2bf(nx1.z, nx1.w);
      *(uint4*)(Xb + xb_b * T_ + xb_u0) = pk;
    }
    __syncthreads();  // writes visible for next iteration
  }
}

extern "C" void kernel_launch(void* const* d_in, const int* in_sizes, int n_in,
                              void* d_out, int out_size, void* d_ws, size_t ws_size,
                              hipStream_t stream) {
  (void)in_sizes; (void)n_in; (void)d_ws; (void)ws_size; (void)out_size;
  const float* xg  = (const float*)d_in[0];  // [B,C,L]
  const float* ldt = (const float*)d_in[1];  // [C]
  const float* arl = (const float*)d_in[2];  // [C,S]
  const float* aim = (const float*)d_in[3];  // [C,S]
  const float* cin = (const float*)d_in[4];  // [C,S,2]
  float* out = (float*)d_out;                // [B,C,L]
  s4d_chunkscan_wmma<<<dim3(C_), dim3(128), 0, stream>>>(xg, ldt, arl, aim, cin, out);
}
